// DotProductAttention_40699110097731
// MI455X (gfx1250) — matrix operand order repaired and checked by reading, not compiled, not run on hardware
//
#include <hip/hip_runtime.h>

typedef __attribute__((ext_vector_type(16))) __bf16 v16bf;
typedef __attribute__((ext_vector_type(8)))  float  v8f;

#define NEG_L2E  (-1442695.04f)                     // -1e6 * log2(e)
#define QSCALE   (0.125f * 1.4426950408889634f)     // 1/sqrt(64) * log2(e)

static __device__ __forceinline__ int   f2i(float x){ union{float f;int i;}u; u.f=x; return u.i; }
static __device__ __forceinline__ float i2f(int   x){ union{int i;float f;}u; u.i=x; return u.f; }

// 16-lane-half butterfly steps on the VALU pipe (no LDS traffic).
static __device__ __forceinline__ float dpp_xor1(float x){
  return i2f(__builtin_amdgcn_mov_dpp8(f2i(x),
        (1<<0)|(0<<3)|(3<<6)|(2<<9)|(5<<12)|(4<<15)|(7<<18)|(6<<21)));
}
static __device__ __forceinline__ float dpp_xor2(float x){
  return i2f(__builtin_amdgcn_mov_dpp8(f2i(x),
        (2<<0)|(3<<3)|(0<<6)|(1<<9)|(6<<12)|(7<<15)|(4<<18)|(5<<21)));
}
static __device__ __forceinline__ float dpp_xor4(float x){
  return i2f(__builtin_amdgcn_mov_dpp8(f2i(x),
        (4<<0)|(5<<3)|(6<<6)|(7<<9)|(0<<12)|(1<<15)|(2<<18)|(3<<21)));
}
static __device__ __forceinline__ float dpp_xor8(float x){
  const int i = f2i(x);
  // DPP16 row_ror:8 -- rotate by 8 within a 16-lane row == lane ^ 8
  return i2f(__builtin_amdgcn_update_dpp(i, i, 0x128, 0xf, 0xf, true));
}
static __device__ __forceinline__ float hmax16(float x){
  x = fmaxf(x, dpp_xor1(x)); x = fmaxf(x, dpp_xor2(x));
  x = fmaxf(x, dpp_xor4(x)); x = fmaxf(x, dpp_xor8(x));
  return x;
}
static __device__ __forceinline__ float hsum16(float x){
  x += dpp_xor1(x); x += dpp_xor2(x); x += dpp_xor4(x); x += dpp_xor8(x);
  return x;
}

static __device__ __forceinline__ v8f bwmma(const v16bf a, const v16bf b, const v8f c){
  return __builtin_amdgcn_wmma_f32_16x16x32_bf16(false, a, false, b, (short)0, c, false, false);
}

union frag16 { v16bf v; uint4 q[2]; };

// One 64-key tile: scores (8 WMMA) + online softmax + PV (8 WMMA).
template <bool MASK>
static __device__ __forceinline__ void process_tile(
    int kv0, int L, int ln, int hi,
    const v16bf aQ0, const v16bf aQ1,
    const __bf16 (*__restrict__ Ksp)[72],
    const __bf16 (*__restrict__ Vtp)[72],
    __bf16 (*__restrict__ Pwp)[72],
    float m_i[8], float l_i[8], v8f oacc[4])
{
    const int fb = hi ? 8 : 0;      // B 32x16 layout: element j -> k = fb + (j>>3)*16 + (j&7) (+32 per chunk)

    // ---- preload ALL score B-fragments, then run WMMAs back-to-back ----
    frag16 bk[8];
    #pragma unroll
    for (int sub = 0; sub < 4; sub++) {
        const uint4* p0 = (const uint4*)&Ksp[sub * 16 + ln][fb];
        const uint4* p1 = (const uint4*)&Ksp[sub * 16 + ln][fb + 32];
        bk[2 * sub    ].q[0] = p0[0]; bk[2 * sub    ].q[1] = *(const uint4*)&Ksp[sub * 16 + ln][fb + 16];
        bk[2 * sub + 1].q[0] = p1[0]; bk[2 * sub + 1].q[1] = *(const uint4*)&Ksp[sub * 16 + ln][fb + 48];
    }
    v8f c[4];
    #pragma unroll
    for (int sub = 0; sub < 4; sub++) {
        v8f acc = {};
        acc = bwmma(aQ0, bk[2 * sub    ].v, acc);
        acc = bwmma(aQ1, bk[2 * sub + 1].v, acc);
        c[sub] = acc;
    }

    // ---- issue V B-fragment loads now; latency hides under the softmax ----
    frag16 bv[8];
    #pragma unroll
    for (int s = 0; s < 4; s++) {
        const uint4* pv0 = (const uint4*)&Vtp[s * 16 + ln][fb];
        const uint4* pv1 = (const uint4*)&Vtp[s * 16 + ln][fb + 32];
        bv[2 * s    ].q[0] = pv0[0]; bv[2 * s    ].q[1] = *(const uint4*)&Vtp[s * 16 + ln][fb + 16];
        bv[2 * s + 1].q[0] = pv1[0]; bv[2 * s + 1].q[1] = *(const uint4*)&Vtp[s * 16 + ln][fb + 48];
    }

    // ---- online softmax (log2 domain; exp2 only) ----
    float sc[8];
    #pragma unroll
    for (int r = 0; r < 8; r++) {
        float s0 = c[0][r], s1 = c[1][r], s2 = c[2][r], s3 = c[3][r];
        if (MASK) {
            const int n = kv0 + ln;
            s0 = (n      < L) ? s0 : NEG_L2E;
            s1 = (n + 16 < L) ? s1 : NEG_L2E;
            s2 = (n + 32 < L) ? s2 : NEG_L2E;
            s3 = (n + 48 < L) ? s3 : NEG_L2E;
        }
        float mx = fmaxf(fmaxf(s0, s1), fmaxf(s2, s3));
        mx = hmax16(mx);
        const float mN = fmaxf(m_i[r], mx);
        const float p0 = exp2f(s0 - mN);
        const float p1 = exp2f(s1 - mN);
        const float p2 = exp2f(s2 - mN);
        const float p3 = exp2f(s3 - mN);
        const float rs = hsum16((p0 + p1) + (p2 + p3));
        const float f  = exp2f(m_i[r] - mN);
        l_i[r] = l_i[r] * f + rs;
        m_i[r] = mN;
        sc[r]  = f;
        __bf16* prow = &Pwp[r + 8 * hi][ln];
        prow[0]  = (__bf16)p0;
        prow[16] = (__bf16)p1;
        prow[32] = (__bf16)p2;
        prow[48] = (__bf16)p3;
    }
    #pragma unroll
    for (int s = 0; s < 4; s++)
        #pragma unroll
        for (int r = 0; r < 8; r++)
            oacc[s][r] *= sc[r];

    // make this wave's P stores visible to its ds loads (bv already landed)
    asm volatile("s_wait_dscnt 0" ::: "memory");

    // ---- reload P as two A fragments (keys 0..31 and 32..63) ----
    const int kb = hi ? 8 : 0;
    frag16 a0, a1;
    a0.q[0] = *(const uint4*)&Pwp[ln][kb];
    a0.q[1] = *(const uint4*)&Pwp[ln][kb + 16];
    a1.q[0] = *(const uint4*)&Pwp[ln][kb + 32];
    a1.q[1] = *(const uint4*)&Pwp[ln][kb + 48];

    // ---- O += P(16x64) x V(64x64): per d-slice, 2 chained WMMAs ----
    #pragma unroll
    for (int s = 0; s < 4; s++) {
        oacc[s] = bwmma(a0.v, bv[2 * s    ].v, oacc[s]);
        oacc[s] = bwmma(a1.v, bv[2 * s + 1].v, oacc[s]);
    }
}

__launch_bounds__(128)
__global__ void fa_wmma_kernel(const float* __restrict__ Q,
                               const float* __restrict__ K,
                               const float* __restrict__ V,
                               const int*   __restrict__ vlen,
                               float* __restrict__ O)
{
    constexpr int Lq = 2048, Lk = 2048, D = 64;

    __shared__ __bf16 Ks[64][72];      // [key][feat], padded
    __shared__ __bf16 Vt[64][72];      // [feat][key], padded (transposed)
    __shared__ __bf16 Pb[4][16][72];   // per-wave P (16 q rows x 64 keys), padded

    const int tid  = threadIdx.x;
    const int w    = tid >> 5;
    const int lane = tid & 31;
    const int hi   = lane >> 4;
    const int ln   = lane & 15;

    const int b  = blockIdx.x >> 5;    // 32 q-tiles per batch
    const int qt = blockIdx.x & 31;
    const int m0 = qt * 64 + w * 16;   // this wave's 16 query rows

    // ---- Q A-fragments (bf16, 1/sqrt(d)*log2e folded in) ----
    v16bf aQ0, aQ1;
    {
        union { v16bf v; __bf16 h[16]; } u0, u1;
        const float* qrow = Q + ((size_t)b * Lq + m0 + ln) * D;
        const int kb = hi ? 8 : 0;
        #pragma unroll
        for (int j = 0; j < 16; j++) {
            const int k = kb + ((j >> 3) << 4) + (j & 7);
            u0.h[j] = (__bf16)(qrow[k]      * QSCALE);
            u1.h[j] = (__bf16)(qrow[k + 32] * QSCALE);
        }
        aQ0 = u0.v; aQ1 = u1.v;
    }

    const int L = vlen[b];
    int kend;
    if (L <= 0) {
        kend = Lk;                      // all-masked: uniform softmax over all keys
    } else {
        kend = (L + 63) & ~63;
        if (kend > Lk) kend = Lk;
    }

    float m_i[8], l_i[8];
    #pragma unroll
    for (int r = 0; r < 8; r++) { m_i[r] = -3.0e38f; l_i[r] = 0.0f; }
    v8f oacc[4] = {};

    for (int kv0 = 0; kv0 < kend; kv0 += 64) {
        __syncthreads();
        // ---- stage K row-major + V transposed, f32 -> bf16 ----
        {
            const int row = tid >> 1;           // key 0..63
            const int f0  = (tid & 1) << 5;     // 0 or 32
            const float* ks = K + ((size_t)b * Lk + kv0 + row) * D + f0;
            const float* vs = V + ((size_t)b * Lk + kv0 + row) * D + f0;
            #pragma unroll
            for (int i = 0; i < 32; i++) Ks[row][f0 + i] = (__bf16)ks[i];
            #pragma unroll
            for (int i = 0; i < 32; i++) Vt[f0 + i][row] = (__bf16)vs[i];
        }
        __syncthreads();

        // prefetch next tile (uniform branch)
        if (kv0 + 64 < kend) {
            const size_t nb = ((size_t)b * Lk + kv0 + 64 + (tid >> 1)) * D + ((tid & 1) << 5);
            __builtin_prefetch(K + nb, 0, 0);
            __builtin_prefetch(V + nb, 0, 0);
        }

        if (kv0 + 64 <= L)   // wave-uniform: interior tiles need no masking
            process_tile<false>(kv0, L, ln, hi, aQ0, aQ1, Ks, Vt, Pb[w], m_i, l_i, oacc);
        else
            process_tile<true >(kv0, L, ln, hi, aQ0, aQ1, Ks, Vt, Pb[w], m_i, l_i, oacc);
    }

    // ---- epilogue ----
    #pragma unroll
    for (int s = 0; s < 4; s++) {
        #pragma unroll
        for (int r = 0; r < 8; r++) {
            const int m = m0 + r + 8 * hi;
            O[((size_t)b * Lq + m) * D + s * 16 + ln] = oacc[s][r] / l_i[r];
        }
    }
}

extern "C" void kernel_launch(void* const* d_in, const int* in_sizes, int n_in,
                              void* d_out, int out_size, void* d_ws, size_t ws_size,
                              hipStream_t stream) {
    (void)in_sizes; (void)n_in; (void)out_size; (void)d_ws; (void)ws_size;
    const float* Q  = (const float*)d_in[0];
    const float* K  = (const float*)d_in[1];
    const float* V  = (const float*)d_in[2];
    const int*   vl = (const int*)d_in[3];
    float* O = (float*)d_out;

    const int B = 16, Lq = 2048;
    dim3 grid(B * (Lq / 64));   // 512 blocks
    dim3 block(128);            // 4 waves; each wave owns 16 query rows
    hipLaunchKernelGGL(fa_wmma_kernel, grid, block, 0, stream, Q, K, V, vl, O);
}
